// GraphAttentionLayer_53609781789059
// MI455X (gfx1250) — matrix operand order repaired and checked by reading, not compiled, not run on hardware
//
#include <hip/hip_runtime.h>
#include <hip/hip_bf16.h>

typedef __bf16 bf16;
typedef __attribute__((ext_vector_type(16))) __bf16 v16bf;
typedef __attribute__((ext_vector_type(8)))  __bf16 v8bf;
typedef __attribute__((ext_vector_type(8)))  float  v8f;
typedef __attribute__((ext_vector_type(4)))  float  v4f;
typedef __attribute__((ext_vector_type(4)))  int    v4i;

#define N_NODES 8192
#define F_IN    256
#define F_OUT   128
#define ALPHA   0.2f

// ---------------------------------------------------------------------------
// Kernel 1: fp32 -> bf16 conversions. x stays row-major (WMMA A operand),
// W is transposed to 128x256 so WMMA B loads are contiguous.
// ---------------------------------------------------------------------------
__global__ __launch_bounds__(256) void cvt_kernel(const float* __restrict__ x,
                                                  const float* __restrict__ W,
                                                  bf16* __restrict__ xb,
                                                  bf16* __restrict__ Wt) {
    int idx = blockIdx.x * blockDim.x + threadIdx.x;
    int stride = gridDim.x * blockDim.x;
    for (int i = idx; i < N_NODES * F_IN; i += stride)
        xb[i] = (bf16)x[i];
    for (int i = idx; i < F_IN * F_OUT; i += stride) {
        int k = i / F_OUT, n = i % F_OUT;
        Wt[n * F_IN + k] = (bf16)W[i];
    }
}

// ---------------------------------------------------------------------------
// Kernel 2: h = x @ W via v_wmma_f32_16x16x32_bf16.
// Grid: 512 row-tiles (16 rows each), 8 waves/block, wave w owns cols [16w,16w+16).
// Writes h fp32 row-major (for src/dst + output math) and hT bf16 (128 x 8192,
// column-major h) so the attention GEMM's B loads are contiguous.
// ---------------------------------------------------------------------------
__global__ __launch_bounds__(256) void gemm1_kernel(const bf16* __restrict__ xb,
                                                    const bf16* __restrict__ Wt,
                                                    float* __restrict__ h,
                                                    bf16* __restrict__ hT) {
    const int lane = threadIdx.x & 31;
    const int wave = threadIdx.x >> 5;
    const int m0 = blockIdx.x * 16;
    const int n0 = wave * 16;
    const int r15 = lane & 15;      // A row / B col within tile
    const int hi  = lane >> 4;      // lane half selects K sub-range

    v8f acc = {};
    // A: lane half 0 covers K {0..7,16..23}+kb, half 1 covers {8..15,24..31}+kb
    const bf16* aBase = xb + (size_t)(m0 + r15) * F_IN + hi * 8;
    // B: lane half 0 covers K {0..7,16..23}+kb, half 1 covers {8..15,24..31}+kb
    const bf16* bBase = Wt + (size_t)(n0 + r15) * F_IN + hi * 8;

    for (int kb = 0; kb < F_IN; kb += 32) {
        v8bf a_lo = *(const v8bf*)(aBase + kb);
        v8bf a_hi = *(const v8bf*)(aBase + kb + 16);
        v8bf b_lo = *(const v8bf*)(bBase + kb);
        v8bf b_hi = *(const v8bf*)(bBase + kb + 16);
        v16bf A, B;
#pragma unroll
        for (int e = 0; e < 8; ++e) {
            A[e] = a_lo[e]; A[e + 8] = a_hi[e];
            B[e] = b_lo[e]; B[e + 8] = b_hi[e];
        }
        acc = __builtin_amdgcn_wmma_f32_16x16x32_bf16(false, A, false, B,
                                                      (short)0, acc, false, false);
    }
    // C/D layout: VGPR r, lane l -> row = r + 8*(l>=16), col = l%16
#pragma unroll
    for (int r = 0; r < 8; ++r) {
        int row = r + 8 * hi;
        float v = acc[r];
        h[(size_t)(m0 + row) * F_OUT + (n0 + r15)] = v;
        hT[(size_t)(n0 + r15) * N_NODES + (m0 + row)] = (bf16)v;
    }
}

// ---------------------------------------------------------------------------
// Kernel 3: src = h @ a[:128], dst = h @ a[128:]. One wave per row.
// ---------------------------------------------------------------------------
__global__ __launch_bounds__(256) void srcdst_kernel(const float* __restrict__ h,
                                                     const float* __restrict__ a,
                                                     float* __restrict__ src,
                                                     float* __restrict__ dst) {
    const int lane = threadIdx.x & 31;
    const int wave = threadIdx.x >> 5;
    const int row = blockIdx.x * 8 + wave;
    float s1 = 0.f, s2 = 0.f;
    for (int c = lane; c < F_OUT; c += 32) {
        float hv = h[(size_t)row * F_OUT + c];
        s1 += hv * a[c];
        s2 += hv * a[F_OUT + c];
    }
#pragma unroll
    for (int off = 16; off; off >>= 1) {
        s1 += __shfl_xor(s1, off, 32);
        s2 += __shfl_xor(s2, off, 32);
    }
    if (lane == 0) { src[row] = s1; dst[row] = s2; }
}

// ---------------------------------------------------------------------------
// Kernel 4: fused masked-softmax attention + (attention @ h), single pass
// over adj (256 MB read once from HBM -> bandwidth bound, ~11 us roofline).
// exp taken without max-subtraction (|e| ~ 3, shift-invariant softmax).
// One block (8 waves) per 16-row block; each wave owns strided j-chunks of 32,
// accumulates O (16x128 fp32 in 8 WMMA accumulators) and row sums, then
// reduces across waves in LDS with ds_add_f32.
// ---------------------------------------------------------------------------
__global__ __launch_bounds__(256) void attn_kernel(const int* __restrict__ adj,
                                                   const bf16* __restrict__ hT,
                                                   const float* __restrict__ src,
                                                   const float* __restrict__ dstv,
                                                   float* __restrict__ out) {
    __shared__ float Osh[16 * F_OUT];
    __shared__ float Ssh[16];
    const int tid  = threadIdx.x;
    const int lane = tid & 31;
    const int wave = tid >> 5;
    for (int i = tid; i < 16 * F_OUT; i += 256) Osh[i] = 0.f;
    if (tid < 16) Ssh[tid] = 0.f;
    __syncthreads();

    const int m0   = blockIdx.x * 16;
    const int r15  = lane & 15;          // attention row this lane feeds
    const int hi   = lane >> 4;
    const int koff = hi * 8;             // A-matrix K sub-offset for this half
    const float srcv = src[m0 + r15];

    v8f acc[8];
#pragma unroll
    for (int t = 0; t < 8; ++t) acc[t] = (v8f){};
    float ssum = 0.f;

    const int* adjRow = adj + (size_t)(m0 + r15) * N_NODES;

    for (int jb = wave * 32; jb < N_NODES; jb += 8 * 32) {
        if (jb + 8 * 32 < N_NODES)       // uniform branch: stream next adj chunk
            __builtin_prefetch(adjRow + jb + 8 * 32 + koff, 0, 0);

        // adj + dst for this lane's 16 j's: runs [koff..+7] and [koff+16..+23]
        v4i q0 = *(const v4i*)(adjRow + jb + koff);
        v4i q1 = *(const v4i*)(adjRow + jb + koff + 4);
        v4i q2 = *(const v4i*)(adjRow + jb + koff + 16);
        v4i q3 = *(const v4i*)(adjRow + jb + koff + 20);
        v4f d0 = *(const v4f*)(dstv + jb + koff);
        v4f d1 = *(const v4f*)(dstv + jb + koff + 4);
        v4f d2 = *(const v4f*)(dstv + jb + koff + 16);
        v4f d3 = *(const v4f*)(dstv + jb + koff + 20);

        v16bf A;
#pragma unroll
        for (int e = 0; e < 8; ++e) {
            int   av = (e < 4) ? q0[e] : q1[e - 4];
            float dv = (e < 4) ? d0[e] : d1[e - 4];
            float ee = srcv + dv;
            ee = fmaxf(ee, 0.f) + ALPHA * fminf(ee, 0.f);   // leaky relu
            float ex = __expf(ee);
            float p  = (av > 0) ? ex : 0.f;                 // mask (select, no branch)
            ssum += p;
            A[e] = (bf16)p;
        }
#pragma unroll
        for (int e = 0; e < 8; ++e) {
            int   av = (e < 4) ? q2[e] : q3[e - 4];
            float dv = (e < 4) ? d2[e] : d3[e - 4];
            float ee = srcv + dv;
            ee = fmaxf(ee, 0.f) + ALPHA * fminf(ee, 0.f);
            float ex = __expf(ee);
            float p  = (av > 0) ? ex : 0.f;
            ssum += p;
            A[e + 8] = (bf16)p;
        }

        // B tiles from hT (L2-resident, contiguous 16B loads), 8 WMMAs
        const bf16* hTbase = hT + (size_t)r15 * N_NODES + jb + hi * 8;
#pragma unroll
        for (int t = 0; t < 8; ++t) {
            const bf16* bp = hTbase + (size_t)t * 16 * N_NODES;
            v8bf b_lo = *(const v8bf*)(bp);
            v8bf b_hi = *(const v8bf*)(bp + 16);
            v16bf B;
#pragma unroll
            for (int e = 0; e < 8; ++e) { B[e] = b_lo[e]; B[e + 8] = b_hi[e]; }
            acc[t] = __builtin_amdgcn_wmma_f32_16x16x32_bf16(false, A, false, B,
                                                             (short)0, acc[t], false, false);
        }
    }

    // cross-wave reduction in LDS (ds_add_f32)
    atomicAdd(&Ssh[r15], ssum);
#pragma unroll
    for (int t = 0; t < 8; ++t) {
#pragma unroll
        for (int r = 0; r < 8; ++r) {
            int row = r + 8 * hi;
            int col = t * 16 + r15;
            atomicAdd(&Osh[row * F_OUT + col], acc[t][r]);
        }
    }
    __syncthreads();

    for (int i = tid; i < 16 * F_OUT; i += 256) {
        int row = i >> 7, col = i & 127;
        out[(size_t)(m0 + row) * F_OUT + col] = Osh[i] / Ssh[row];
    }
}

// ---------------------------------------------------------------------------
extern "C" void kernel_launch(void* const* d_in, const int* in_sizes, int n_in,
                              void* d_out, int out_size, void* d_ws, size_t ws_size,
                              hipStream_t stream) {
    const float* x   = (const float*)d_in[0];   // 8192 x 256
    const int*   adj = (const int*)d_in[1];     // 8192 x 8192
    const float* W   = (const float*)d_in[2];   // 256 x 128
    const float* a   = (const float*)d_in[3];   // 256 x 1

    char* ws = (char*)d_ws;
    float* h   = (float*)(ws);                              // 4 MB: 8192x128 f32
    bf16*  hT  = (bf16*) (ws + (4u  << 20));                // 2 MB: 128x8192 bf16
    bf16*  xb  = (bf16*) (ws + (6u  << 20));                // 4 MB: 8192x256 bf16
    bf16*  Wt  = (bf16*) (ws + (10u << 20));                // 64 KB: 128x256 bf16
    float* src = (float*)(ws + (10u << 20) + (1u << 16));   // 32 KB
    float* dst = (float*)(ws + (10u << 20) + (1u << 16) + (32u << 10)); // 32 KB

    cvt_kernel   <<<dim3(2048), dim3(256), 0, stream>>>(x, W, xb, Wt);
    gemm1_kernel <<<dim3(N_NODES / 16), dim3(256), 0, stream>>>(xb, Wt, h, hT);
    srcdst_kernel<<<dim3(N_NODES / 8), dim3(256), 0, stream>>>(h, a, src, dst);
    attn_kernel  <<<dim3(N_NODES / 16), dim3(256), 0, stream>>>(adj, hT, src, dst,
                                                                (float*)d_out);
}
